// MSDeformAttn_with_GlobalRegisters_v0_71279277244640
// MI455X (gfx1250) — matrix operand order repaired and checked by reading, not compiled, not run on hardware
//
#include <hip/hip_runtime.h>

// Problem constants (fixed by the reference).
#define NB     2
#define NQ     5506
#define ROWS   11012          // NB*LQ and also NB*LEN_IN
#define CDIM   256
#define NHEAD  8
#define HDIM   32
#define NLVL   4
#define NPTS   4
#define NREG   66
#define RWCOLS 528            // NHEAD*NREG

typedef __attribute__((ext_vector_type(16))) __bf16 v16bf;
typedef __attribute__((ext_vector_type(8)))  float  v8f;

union BF16Frag { v16bf v; uint4 q[2]; unsigned short u[16]; };

__device__ __forceinline__ unsigned short f2bf(float f) {
  unsigned int u = __float_as_uint(f);
  u += 0x7FFFu + ((u >> 16) & 1u);       // round-to-nearest-even
  return (unsigned short)(u >> 16);
}

// Low 32 bits of a generic pointer to LDS = wave-relative LDS byte address
// (the high half picks the LDS aperture; the low half is the LDS offset).
__device__ __forceinline__ unsigned lds_off(const void* p) {
  return (unsigned)(unsigned long long)p;
}

// CDNA5 async copy: global -> LDS, 16B per lane, tracked by ASYNCcnt.
__device__ __forceinline__ void async_load_b128(unsigned lds_byte_addr,
                                                const void* gaddr) {
  asm volatile("global_load_async_to_lds_b128 %0, %1, off"
               :: "v"(lds_byte_addr), "v"(gaddr) : "memory");
}
__device__ __forceinline__ void wait_async0() {
  asm volatile("s_wait_asynccnt 0x0" ::: "memory");
}

// ---------------------------------------------------------------- convert
__global__ void cvt_f32_bf16(const float* __restrict__ src,
                             unsigned short* __restrict__ dst, int n) {
  int i = blockIdx.x * blockDim.x + threadIdx.x;
  if (i < n) dst[i] = f2bf(src[i]);
}

// src: K x Nc row-major fp32 -> dst: Nc x K bf16 (N-major weights for GEMM B)
__global__ void cvt_transpose_bf16(const float* __restrict__ src,
                                   unsigned short* __restrict__ dst,
                                   int K, int Nc) {
  int i = blockIdx.x * blockDim.x + threadIdx.x;
  if (i >= K * Nc) return;
  int k = i / Nc, n = i - k * Nc;
  dst[(size_t)n * K + k] = f2bf(src[i]);
}

// ---------------------------------------------------------------- WMMA GEMM
// C[M x Nc] = A[M x 256] * B^T + bias.
// A  : bf16 row-major, ld = 256 (padded allocation past M rows).
// Bt : bf16 N-major (Nc x 256), padded past Nc rows.
// Tile 128x64, 256 threads = 8 waves (4x2), each wave 2x2 16x16 WMMA tiles.
// Staging: GLOBAL_LOAD_ASYNC_TO_LDS_B128 with double-buffered LDS.
__global__ __launch_bounds__(256)
void wmma_gemm(const unsigned short* __restrict__ A,
               const unsigned short* __restrict__ Bt,
               const float* __restrict__ bias,
               float* __restrict__ C,
               int M, int Nc) {
  __shared__ unsigned short As[2][128 * 32];
  __shared__ unsigned short Bs[2][64 * 32];
  const int K = 256;
  const int tid  = threadIdx.x;
  const int lane = tid & 31;
  const int wave = tid >> 5;
  const int wm = wave & 3, wn = wave >> 2;
  const int m0 = blockIdx.x * 128, n0 = blockIdx.y * 64;

  // Per-thread staging coordinates (fixed across K-steps).
  const int ar = tid >> 1, ach = (tid & 1) * 16;      // A: 2 thr/row, 16 bf16 ea
  const int bn = tid >> 2, bkc = (tid & 3) * 8;       // B: 4 thr/row, 8 bf16 ea

  auto stage = [&](int buf, int k0) {
    const unsigned short* as = A + (size_t)(m0 + ar) * K + k0 + ach;
    unsigned ad = lds_off(&As[buf][ar * 32 + ach]);
    async_load_b128(ad, as);
    async_load_b128(ad + 16, as + 8);
    const unsigned short* bs = Bt + (size_t)(n0 + bn) * K + k0 + bkc;
    async_load_b128(lds_off(&Bs[buf][bn * 32 + bkc]), bs);
  };

  v8f acc[2][2] = {};

  stage(0, 0);
  wait_async0();
  __syncthreads();

  const int mrow = lane & 15, kb = (lane >> 4) * 8;
  const int ncol = lane & 15, ko = (lane >> 4) * 8;

  for (int it = 0; it < 8; ++it) {
    const int buf = it & 1;
    if (it < 7) stage(buf ^ 1, (it + 1) * 32);   // prefetch next tile (async)

    BF16Frag afrag[2], bfrag[2];
#pragma unroll
    for (int t = 0; t < 2; ++t) {
      const unsigned short* rp = &As[buf][(wm * 32 + t * 16 + mrow) * 32];
      afrag[t].q[0] = *(const uint4*)(rp + kb);
      afrag[t].q[1] = *(const uint4*)(rp + 16 + kb);
    }
#pragma unroll
    for (int t = 0; t < 2; ++t) {
      const unsigned short* rp = &Bs[buf][(wn * 32 + t * 16 + ncol) * 32];
      bfrag[t].q[0] = *(const uint4*)(rp + ko);
      bfrag[t].q[1] = *(const uint4*)(rp + 16 + ko);
    }
#pragma unroll
    for (int i = 0; i < 2; ++i)
#pragma unroll
      for (int j = 0; j < 2; ++j)
        acc[i][j] = __builtin_amdgcn_wmma_f32_16x16x32_bf16(
            false, afrag[i].v, false, bfrag[j].v, (short)0, acc[i][j],
            false, false);

    wait_async0();       // next tile resident in LDS
    __syncthreads();     // everyone done reading current buffer
  }

  // ---- epilogue (C/D layout: lanes 0-15 M=vgpr, lanes 16-31 M=8+vgpr; N=lane&15)
#pragma unroll
  for (int i = 0; i < 2; ++i) {
    int rbase = m0 + wm * 32 + i * 16 + 8 * (lane >> 4);
#pragma unroll
    for (int j = 0; j < 2; ++j) {
      int col = n0 + wn * 32 + j * 16 + (lane & 15);
      if (col >= Nc) continue;
      float bv = bias ? bias[col] : 0.f;
#pragma unroll
      for (int v = 0; v < 8; ++v) {
        int row = rbase + v;
        if (row < M) C[(size_t)row * Nc + col] = acc[i][j][v] + bv;
      }
    }
  }
}

// ---------------------------------------------------------------- softmaxes
__global__ void softmax16(float* __restrict__ x, int total) {
  int gid = blockIdx.x * blockDim.x + threadIdx.x;
  if (gid >= total) return;
  int bn = gid >> 3, h = gid & 7;
  float* p = x + (size_t)bn * 128 + h * 16;
  float v[16], m = -3.0e38f;
#pragma unroll
  for (int i = 0; i < 16; ++i) { v[i] = p[i]; m = fmaxf(m, v[i]); }
  float s = 0.f;
#pragma unroll
  for (int i = 0; i < 16; ++i) { v[i] = __expf(v[i] - m); s += v[i]; }
  float r = 1.f / s;
#pragma unroll
  for (int i = 0; i < 16; ++i) p[i] = v[i] * r;
}

__global__ void softmax66(float* __restrict__ x, int total) {
  int gid = blockIdx.x * blockDim.x + threadIdx.x;
  if (gid >= total) return;
  int bn = gid >> 3, h = gid & 7;
  float* p = x + (size_t)bn * RWCOLS + h * NREG;
  float m = -3.0e38f;
  for (int i = 0; i < NREG; ++i) m = fmaxf(m, p[i]);
  float s = 0.f;
  for (int i = 0; i < NREG; ++i) { float e = __expf(p[i] - m); p[i] = e; s += e; }
  float r = 1.f / s;
  for (int i = 0; i < NREG; ++i) p[i] *= r;
}

// ---------------------------------------------------------------- deformable gather
// value: (b, pos, h, d) fp32; hw rows start at pos = NREG.
// 8 threads per (bn, h), each owns 4 channels.
__global__ void deform_sample(const float* __restrict__ value,
                              const float* __restrict__ refpts,
                              const float* __restrict__ offs,
                              const float* __restrict__ attn,
                              float* __restrict__ hw) {
  int gid = blockIdx.x * blockDim.x + threadIdx.x;
  if (gid >= ROWS * NHEAD * 8) return;
  int quad = gid & 7;
  int h    = (gid >> 3) & 7;
  int bn   = gid >> 6;
  int b    = (bn >= NQ) ? 1 : 0;
  int d0   = quad * 4;

  const int HWs[4] = {64, 32, 16, 8};
  const int st[4]  = {0, 4096, 5120, 5376};
  float a0 = 0.f, a1 = 0.f, a2 = 0.f, a3 = 0.f;

#pragma unroll
  for (int l = 0; l < 4; ++l) {
    const int W = HWs[l], H = HWs[l];
    float rx = refpts[((size_t)bn * NLVL + l) * 2 + 0];
    float ry = refpts[((size_t)bn * NLVL + l) * 2 + 1];
    size_t vrow0 = (size_t)(b * NQ) + NREG + st[l];
#pragma unroll
    for (int p = 0; p < 4; ++p) {
      float ox = offs[(size_t)bn * 256 + (((h * 4 + l) * 4 + p) * 2) + 0];
      float oy = offs[(size_t)bn * 256 + (((h * 4 + l) * 4 + p) * 2) + 1];
      float aw = attn[(size_t)bn * 128 + h * 16 + l * 4 + p];
      // loc = ref + off/norm;  x = loc.x*W - 0.5  ==>  x = rx*W + ox - 0.5
      float x = rx * (float)W + ox - 0.5f;
      float y = ry * (float)H + oy - 0.5f;
      float xf = floorf(x), yf = floorf(y);
      float fx = x - xf, fy = y - yf;
      int ix = (int)xf, iy = (int)yf;
#pragma unroll
      for (int dy = 0; dy < 2; ++dy)
#pragma unroll
        for (int dx = 0; dx < 2; ++dx) {
          int xi = ix + dx, yi = iy + dy;
          if (xi >= 0 && xi < W && yi >= 0 && yi < H) {
            float w = (dx ? fx : 1.f - fx) * (dy ? fy : 1.f - fy) * aw;
            size_t row = vrow0 + (size_t)yi * W + xi;
            const float* vp = value + ((row * NHEAD + h) * HDIM + d0);
            float4 vv = *(const float4*)vp;
            a0 += w * vv.x; a1 += w * vv.y; a2 += w * vv.z; a3 += w * vv.w;
          }
        }
    }
  }
  float4 o = make_float4(a0, a1, a2, a3);
  *(float4*)(hw + ((size_t)bn * NHEAD + h) * HDIM + d0) = o;
}

// ---------------------------------------------------------------- reg mix + lambda combine -> bf16 A for final GEMM
__global__ void reg_combine(const float* __restrict__ value,
                            const float* __restrict__ rw,
                            const float* __restrict__ hw,
                            unsigned short* __restrict__ comb) {
  int gid = blockIdx.x * blockDim.x + threadIdx.x;
  if (gid >= ROWS * NHEAD * 8) return;
  int quad = gid & 7;
  int h    = (gid >> 3) & 7;
  int bn   = gid >> 6;
  int b    = (bn >= NQ) ? 1 : 0;
  int d0   = quad * 4;

  const float* rwp = rw + (size_t)bn * RWCOLS + h * NREG;
  float a0 = 0.f, a1 = 0.f, a2 = 0.f, a3 = 0.f;
  for (int r = 0; r < NREG; ++r) {
    float w = rwp[r];
    const float* vp = value + (((size_t)(b * NQ + r)) * NHEAD + h) * HDIM + d0;
    float4 vv = *(const float4*)vp;
    a0 += w * vv.x; a1 += w * vv.y; a2 += w * vv.z; a3 += w * vv.w;
  }
  float4 hv = *(const float4*)(hw + ((size_t)bn * NHEAD + h) * HDIM + d0);
  union { unsigned short s[4]; uint2 v; } o;
  o.s[0] = f2bf(0.5f * hv.x + 0.5f * a0);
  o.s[1] = f2bf(0.5f * hv.y + 0.5f * a1);
  o.s[2] = f2bf(0.5f * hv.z + 0.5f * a2);
  o.s[3] = f2bf(0.5f * hv.w + 0.5f * a3);
  *(uint2*)(comb + (size_t)bn * 256 + h * 32 + d0) = o.v;
}

// ---------------------------------------------------------------- launcher
extern "C" void kernel_launch(void* const* d_in, const int* in_sizes, int n_in,
                              void* d_out, int out_size, void* d_ws, size_t ws_size,
                              hipStream_t stream) {
  (void)in_sizes; (void)n_in; (void)out_size; (void)ws_size;
  const float* query   = (const float*)d_in[0];
  const float* refpts  = (const float*)d_in[1];
  const float* flatten = (const float*)d_in[2];
  const float* W_off   = (const float*)d_in[5];
  const float* b_off   = (const float*)d_in[6];
  const float* W_attn  = (const float*)d_in[7];
  const float* b_attn  = (const float*)d_in[8];
  const float* W_val   = (const float*)d_in[9];
  const float* b_val   = (const float*)d_in[10];
  const float* W_out   = (const float*)d_in[11];
  const float* b_out   = (const float*)d_in[12];
  const float* task_W  = (const float*)d_in[13];
  const float* task_b  = (const float*)d_in[14];
  float* out = (float*)d_out;

  char* ws = (char*)d_ws;
  size_t cur = 0;
  auto alloc = [&](size_t bytes) -> void* {
    cur = (cur + 255) & ~(size_t)255;
    void* p = ws + cur;
    cur += bytes;
    return p;
  };

  const int MPAD = 128 * ((ROWS + 127) / 128);   // 11136, safe async over-read
  unsigned short* q_bf    = (unsigned short*)alloc((size_t)MPAD * CDIM * 2);
  unsigned short* x_bf    = (unsigned short*)alloc((size_t)MPAD * CDIM * 2);
  unsigned short* Wval_bf = (unsigned short*)alloc((size_t)CDIM * CDIM * 2);   // 256x256 N-major
  unsigned short* Woff_bf = (unsigned short*)alloc((size_t)CDIM * CDIM * 2);   // 256x256 N-major
  unsigned short* Watt_bf = (unsigned short*)alloc((size_t)128 * CDIM * 2);    // 128x256 N-major
  unsigned short* Wout_bf = (unsigned short*)alloc((size_t)CDIM * CDIM * 2);   // 256x256 N-major
  unsigned short* Wtsk_bf = (unsigned short*)alloc((size_t)576 * CDIM * 2);    // 528(+pad)x256, already N-major
  float* value  = (float*)alloc((size_t)ROWS * CDIM * 4);
  float* offbuf = (float*)alloc((size_t)ROWS * CDIM * 4);
  float* attbuf = (float*)alloc((size_t)ROWS * 128 * 4);
  float* rwbuf  = (float*)alloc((size_t)ROWS * RWCOLS * 4);
  float* hwbuf  = (float*)alloc((size_t)ROWS * CDIM * 4);
  unsigned short* comb_bf = (unsigned short*)alloc((size_t)MPAD * CDIM * 2);
  (void)alloc(4096);                              // tail guard

  const int T = 256;
  auto g1 = [](int n) { return dim3((n + 255) / 256); };

  // 1) fp32 -> bf16 staging (weights transposed to N-major; task_W already is)
  cvt_f32_bf16<<<g1(ROWS * CDIM), T, 0, stream>>>(query,   q_bf, ROWS * CDIM);
  cvt_f32_bf16<<<g1(ROWS * CDIM), T, 0, stream>>>(flatten, x_bf, ROWS * CDIM);
  cvt_transpose_bf16<<<g1(CDIM * CDIM), T, 0, stream>>>(W_val,  Wval_bf, CDIM, 256);
  cvt_transpose_bf16<<<g1(CDIM * CDIM), T, 0, stream>>>(W_off,  Woff_bf, CDIM, 256);
  cvt_transpose_bf16<<<g1(CDIM * 128),  T, 0, stream>>>(W_attn, Watt_bf, CDIM, 128);
  cvt_transpose_bf16<<<g1(CDIM * CDIM), T, 0, stream>>>(W_out,  Wout_bf, CDIM, 256);
  cvt_f32_bf16<<<g1(RWCOLS * CDIM), T, 0, stream>>>(task_W, Wtsk_bf, RWCOLS * CDIM);

  const int MB = (ROWS + 127) / 128;  // 87

  // 2) projections (WMMA, async-LDS staged)
  wmma_gemm<<<dim3(MB, 4), T, 0, stream>>>(x_bf, Wval_bf, b_val,  value,  ROWS, 256);
  wmma_gemm<<<dim3(MB, 4), T, 0, stream>>>(q_bf, Woff_bf, b_off,  offbuf, ROWS, 256);
  wmma_gemm<<<dim3(MB, 2), T, 0, stream>>>(q_bf, Watt_bf, b_attn, attbuf, ROWS, 128);
  wmma_gemm<<<dim3(MB, 9), T, 0, stream>>>(q_bf, Wtsk_bf, task_b, rwbuf,  ROWS, RWCOLS);

  // 3) softmaxes
  softmax16<<<g1(ROWS * NHEAD), T, 0, stream>>>(attbuf, ROWS * NHEAD);
  softmax66<<<g1(ROWS * NHEAD), T, 0, stream>>>(rwbuf,  ROWS * NHEAD);

  // 4) deformable bilinear gather (hw_info)
  deform_sample<<<g1(ROWS * NHEAD * 8), T, 0, stream>>>(value, refpts, offbuf, attbuf, hwbuf);

  // 5) register mix + lambda combine -> bf16
  reg_combine<<<g1(ROWS * NHEAD * 8), T, 0, stream>>>(value, rwbuf, hwbuf, comb_bf);

  // 6) output projection (WMMA) -> d_out
  wmma_gemm<<<dim3(MB, 4), T, 0, stream>>>(comb_bf, Wout_bf, b_out, out, ROWS, 256);
}
